// KVCacheMoE_44444321579062
// MI455X (gfx1250) — matrix operand order repaired and checked by reading, not compiled, not run on hardware
//
#include <hip/hip_runtime.h>
#include <hip/hip_bf16.h>

// MoE dense-all-experts forward for MI455X (gfx1250), wave32 + WMMA f16 +
// async global->LDS staging (CDNA5 ASYNCcnt path), quad-buffered.
//
// Pipeline:
//   1) mean over seq (2-phase)  -> xm [B,H]
//   2) router MLP + softmax     -> routing [B,E]
//   3) convert x -> f16         -> xh  [B*S, H]        (32 MB ws)
//   4) transpose+convert We     -> wet [E, H(n), H(k)] (16 MB ws)
//   5) fused all-expert GEMM: out[m,n] = sum_e r[b,e]*relu(xh[m,:]@wet[e,n,:]^T + be[e,n])
//      v_wmma_f32_16x16x32_f16, quad-buffered global_load_async_to_lds_b128,
//      incremental address generation, 128x128 block tile, 32x64 wave tile.

#define B_ 8
#define S_ 2048
#define H_ 1024
#define E_ 8

#define MT 128   // M tile (tokens)
#define NT 128   // N tile (output dims)
#define KT 32    // K step (matches wmma f16 K)
#define LDA 40   // padded LDS row stride in halves (80 bytes, conflict-free)
#define NBUF 4   // quad-buffered async staging (prefetch distance 2)
#define BUFB (MT * LDA * 2)    // 10240 bytes per A (=per B) buffer
#define KSTEPS (H_ / KT)       // 32
#define NSTAGE (E_ * KSTEPS)   // 256

#define SCH 16               // seq chunks for 2-phase mean
#define SPER (S_ / SCH)      // 128

typedef __attribute__((ext_vector_type(16))) _Float16 v16h;
typedef __attribute__((ext_vector_type(8)))  float    v8f;
typedef __attribute__((ext_vector_type(4)))  float    v4f;
typedef __attribute__((ext_vector_type(4)))  _Float16 v4h;

union Frag {
    v16h v;
    v4f  f[2];
};

// ---------------------------------------------------------------- mean phase 1
__global__ __launch_bounds__(256) void kvmoe_mean1_kernel(
    const float* __restrict__ x, float* __restrict__ partial)
{
    int idx = blockIdx.x * 256 + threadIdx.x;   // 0 .. B*H-1
    int c   = blockIdx.y;                       // seq chunk
    int b = idx / H_;
    int h = idx - b * H_;
    const float* p = x + (size_t)b * S_ * H_ + (size_t)c * SPER * H_ + h;
    float s = 0.f;
    for (int i = 0; i < SPER; ++i) s += p[(size_t)i * H_];
    partial[(size_t)c * (B_ * H_) + idx] = s;
}

// ---------------------------------------------------------------- mean phase 2
__global__ __launch_bounds__(256) void kvmoe_mean2_kernel(
    const float* __restrict__ partial, float* __restrict__ xm)
{
    int idx = blockIdx.x * 256 + threadIdx.x;
    float s = 0.f;
    for (int c = 0; c < SCH; ++c) s += partial[(size_t)c * (B_ * H_) + idx];
    xm[idx] = s * (1.0f / (float)S_);
}

// ---------------------------------------------------------------- router MLP
__global__ __launch_bounds__(256) void kvmoe_router_kernel(
    const float* __restrict__ xm,
    const float* __restrict__ Wr1, const float* __restrict__ br1,
    const float* __restrict__ Wr2, const float* __restrict__ br2,
    float* __restrict__ routing)
{
    __shared__ float sxm[H_];
    __shared__ float sh[H_];
    __shared__ float slog[E_];
    const int b = blockIdx.x;
    const int t = threadIdx.x;

    for (int j = t; j < H_; j += 256) sxm[j] = xm[b * H_ + j];
    __syncthreads();

    for (int j = t; j < H_; j += 256) {
        float s = br1[j];
        for (int k = 0; k < H_; ++k) s += sxm[k] * Wr1[(size_t)k * H_ + j];
        sh[j] = s > 0.f ? s : 0.f;
    }
    __syncthreads();

    if (t < E_) {
        float s = br2[t];
        for (int j = 0; j < H_; ++j) s += sh[j] * Wr2[(size_t)j * E_ + t];
        slog[t] = s;
    }
    __syncthreads();

    if (t == 0) {
        float mx = slog[0];
        for (int e = 1; e < E_; ++e) mx = slog[e] > mx ? slog[e] : mx;
        float ex[E_];
        float sum = 0.f;
        for (int e = 0; e < E_; ++e) { ex[e] = __expf(slog[e] - mx); sum += ex[e]; }
        float inv = 1.0f / sum;
        for (int e = 0; e < E_; ++e) routing[b * E_ + e] = ex[e] * inv;
    }
}

// ---------------------------------------------------------------- x -> f16
__global__ __launch_bounds__(256) void kvmoe_cvt_x_kernel(
    const float* __restrict__ x, _Float16* __restrict__ xh)
{
    size_t i4 = (size_t)blockIdx.x * 256 + threadIdx.x;  // one float4 per thread
    v4f v = reinterpret_cast<const v4f*>(x)[i4];
    v4h h;
    h.x = (_Float16)v.x; h.y = (_Float16)v.y;
    h.z = (_Float16)v.z; h.w = (_Float16)v.w;
    reinterpret_cast<v4h*>(xh)[i4] = h;
}

// ------------------------------------------------- We [E,K,N] -> wet [E,N,K] f16
__global__ __launch_bounds__(256) void kvmoe_tr_we_kernel(
    const float* __restrict__ We, _Float16* __restrict__ wet)
{
    __shared__ _Float16 tile[32][33];
    const int e  = blockIdx.z;
    const int n0 = blockIdx.x * 32;
    const int k0 = blockIdx.y * 32;
    const int tx = threadIdx.x & 31;
    const int ty = threadIdx.x >> 5;

    const float* src = We + (size_t)e * H_ * H_;
    for (int r = ty; r < 32; r += 8)
        tile[r][tx] = (_Float16)src[(size_t)(k0 + r) * H_ + n0 + tx];
    __syncthreads();
    _Float16* dst = wet + (size_t)e * H_ * H_;
    for (int r = ty; r < 32; r += 8)
        dst[(size_t)(n0 + r) * H_ + k0 + tx] = tile[tx][r];
}

// ---------------------------------------------------------------- fused MoE GEMM
// grid: (H/NT, B*S/MT), block: 256 threads = 8 waves (wave32).
// Block tile 128x128; wave w covers 32 (M) x 64 (N): mw=(w&3)*32, nw=(w>>2)*64,
// i.e. 2x4 subtiles of 16x16 -> 8 WMMAs per wave per K-step.
// Staging: global_load_async_to_lds_b128 (ASYNCcnt), 4 buffers, prefetch dist 2,
// one barrier per stage, incremental global pointers (+64 B/stage).
__global__ __launch_bounds__(256) void kvmoe_gemm_kernel(
    const _Float16* __restrict__ xh,      // [B*S, H]
    const _Float16* __restrict__ wet,     // [E, H(n), H(k)]
    const float* __restrict__ be,         // [E, H]
    const float* __restrict__ routing,    // [B, E]
    float* __restrict__ out)              // [B*S, H]
{
    __shared__ alignas(16) _Float16 Als[NBUF][MT * LDA];   // 4 x 10 KB
    __shared__ alignas(16) _Float16 Bls[NBUF][NT * LDA];   // 4 x 10 KB (80 KB total)

    const int n0 = blockIdx.x * NT;
    const int m0 = blockIdx.y * MT;
    const int b  = m0 / S_;               // tile lies in one batch (128 | 2048)
    const int t  = threadIdx.x;
    const int w  = t >> 5;
    const int lane  = t & 31;
    const int lhalf = lane >> 4;
    const int l16   = lane & 15;

    const int mw = (w & 3) * 32;          // wave M offset within tile
    const int nw = (w >> 2) * 64;         // wave N offset within tile

    // WMMA 16-bit fragment K offsets (halves) per ISA layout
    const int kcA = lhalf * 8;            // A: lanes<16 -> K {0..7,16..23}
    const int kcB = lhalf * 8;            // B: lanes<16 -> K {0..7,16..23}

    // ---- staging: each thread owns one row (t>>1) and two adjacent 16B chunks
    // at byte offsets +0/+16 within that row, for both A and B tiles.
    // async IOFFSET is added to BOTH lds and global addresses (ISA 08 §4.4),
    // so one address pair + offset:16 covers the second chunk.
    const int srow  = t >> 1;
    const int sboff = (t & 1) * 32;       // 0 or 32 bytes within the 64B row

    uint64_t gA = (uint64_t)(uintptr_t)xh
                + ((uint64_t)(m0 + srow) * H_) * 2 + (uint64_t)sboff;
    uint64_t gB = (uint64_t)(uintptr_t)wet
                + ((uint64_t)(n0 + srow) * H_) * 2 + (uint64_t)sboff;
    uint32_t laA = (uint32_t)(uintptr_t)&Als[0][0] + srow * (LDA * 2) + sboff;
    uint32_t laB = (uint32_t)(uintptr_t)&Bls[0][0] + srow * (LDA * 2) + sboff;

    // Issue stage q (4 async b128 per thread) into buffer q&3, then advance
    // pointers for q+1 (+64B; expert rollover: A -1984B, B +2*H*H-1984B).
    auto stage = [&](int q) {
        const uint32_t boff = (uint32_t)(q & (NBUF - 1)) * BUFB;
        uint32_t a0 = laA + boff;
        uint32_t b0 = laB + boff;
        asm volatile("global_load_async_to_lds_b128 %0, %1, off"
                     :: "v"(a0), "v"(gA) : "memory");
        asm volatile("global_load_async_to_lds_b128 %0, %1, off offset:16"
                     :: "v"(a0), "v"(gA) : "memory");
        asm volatile("global_load_async_to_lds_b128 %0, %1, off"
                     :: "v"(b0), "v"(gB) : "memory");
        asm volatile("global_load_async_to_lds_b128 %0, %1, off offset:16"
                     :: "v"(b0), "v"(gB) : "memory");
        if (((q + 1) & (KSTEPS - 1)) == 0) {   // next stage starts a new expert
            gA -= (uint64_t)((KSTEPS - 1) * KT * 2);                    // k: 992->0
            gB += (uint64_t)(2u * H_ * H_) - (uint64_t)((KSTEPS - 1) * KT * 2);
        } else {
            gA += KT * 2;
            gB += KT * 2;
        }
    };

    v8f outacc[2][4];
    for (int i = 0; i < 2; ++i)
        for (int j = 0; j < 4; ++j)
            outacc[i][j] = {};

    // Prologue: prefetch stages 0 and 1.
    stage(0);
    stage(1);

    for (int e = 0; e < E_; ++e) {
        v8f acc[2][4];
        for (int i = 0; i < 2; ++i)
            for (int j = 0; j < 4; ++j)
                acc[i][j] = {};

        for (int kk = 0; kk < KSTEPS; ++kk) {
            const int s  = (e << 5) + kk;
            const int bi = s & (NBUF - 1);

            // Prefetch stage s+2 (buffer reuse distance 4 > prefetch dist 2;
            // the one-barrier-per-stage chain makes this race-free).
            if (s + 2 < NSTAGE) stage(s + 2);

            // 4 async instructions per stage per wave; allow the newest
            // <=2 stages to remain in flight => stage s has landed.
            if (s + 2 < NSTAGE)
                asm volatile("s_wait_asynccnt 0x8" ::: "memory");
            else if (s + 1 < NSTAGE)
                asm volatile("s_wait_asynccnt 0x4" ::: "memory");
            else
                asm volatile("s_wait_asynccnt 0x0" ::: "memory");

            __syncthreads();   // all waves' stage-s data visible in LDS

            // ---- load fragments from LDS buffer bi
            const _Float16* Ab = &Als[bi][0];
            const _Float16* Bb = &Bls[bi][0];
            Frag a[2], bf[4];
#pragma unroll
            for (int i = 0; i < 2; ++i) {
                int r = mw + i * 16 + l16;
                a[i].f[0] = *reinterpret_cast<const v4f*>(&Ab[r * LDA + kcA]);
                a[i].f[1] = *reinterpret_cast<const v4f*>(&Ab[r * LDA + kcA + 16]);
            }
#pragma unroll
            for (int j = 0; j < 4; ++j) {
                int r = nw + j * 16 + l16;
                bf[j].f[0] = *reinterpret_cast<const v4f*>(&Bb[r * LDA + kcB]);
                bf[j].f[1] = *reinterpret_cast<const v4f*>(&Bb[r * LDA + kcB + 16]);
            }

            // ---- 8 WMMAs per wave per stage
#pragma unroll
            for (int i = 0; i < 2; ++i)
#pragma unroll
                for (int j = 0; j < 4; ++j)
                    acc[i][j] = __builtin_amdgcn_wmma_f32_16x16x32_f16(
                        false, a[i].v, false, bf[j].v,
                        (short)0, acc[i][j], false, false);
        }

        // ---- expert epilogue: bias + relu + routing weight, accumulate
        const float rw = routing[b * E_ + e];
#pragma unroll
        for (int j = 0; j < 4; ++j) {
            const float bv = be[e * H_ + n0 + nw + j * 16 + l16];
#pragma unroll
            for (int i = 0; i < 2; ++i) {
#pragma unroll
                for (int r = 0; r < 8; ++r) {
                    float v = acc[i][j][r] + bv;
                    v = v > 0.f ? v : 0.f;
                    outacc[i][j][r] += rw * v;
                }
            }
        }
    }

    // ---- store C fragments: VGPR r, lanes<16 -> row r, lanes>=16 -> row r+8
#pragma unroll
    for (int i = 0; i < 2; ++i)
#pragma unroll
        for (int j = 0; j < 4; ++j)
#pragma unroll
            for (int r = 0; r < 8; ++r) {
                int row = m0 + mw + i * 16 + r + lhalf * 8;
                int col = n0 + nw + j * 16 + l16;
                out[(size_t)row * H_ + col] = outacc[i][j][r];
            }
}

extern "C" void kernel_launch(void* const* d_in, const int* in_sizes, int n_in,
                              void* d_out, int out_size, void* d_ws, size_t ws_size,
                              hipStream_t stream) {
    const float* x   = (const float*)d_in[0];
    const float* We  = (const float*)d_in[1];
    const float* be  = (const float*)d_in[2];
    const float* Wr1 = (const float*)d_in[3];
    const float* br1 = (const float*)d_in[4];
    const float* Wr2 = (const float*)d_in[5];
    const float* br2 = (const float*)d_in[6];
    float* out = (float*)d_out;

    char* ws = (char*)d_ws;
    _Float16* xh  = (_Float16*)ws;                                   // 32 MB
    _Float16* wet = (_Float16*)(ws + (size_t)B_ * S_ * H_ * 2);      // 16 MB
    float* xm      = (float*)(ws + (size_t)B_ * S_ * H_ * 2
                                 + (size_t)E_ * H_ * H_ * 2);        // 32 KB
    float* routing = xm + B_ * H_;                                   // 256 B
    float* partial = routing + B_ * E_;                              // 512 KB

    // router path (tiny)
    kvmoe_mean1_kernel<<<dim3((B_ * H_) / 256, SCH), 256, 0, stream>>>(x, partial);
    kvmoe_mean2_kernel<<<dim3((B_ * H_) / 256), 256, 0, stream>>>(partial, xm);
    kvmoe_router_kernel<<<dim3(B_), 256, 0, stream>>>(xm, Wr1, br1, Wr2, br2, routing);

    // precision conversion / layout
    kvmoe_cvt_x_kernel<<<dim3((B_ * S_ * H_) / (256 * 4)), 256, 0, stream>>>(x, xh);
    kvmoe_tr_we_kernel<<<dim3(H_ / 32, H_ / 32, E_), 256, 0, stream>>>(We, wet);

    // fused all-expert GEMM + relu + routing-weighted sum
    kvmoe_gemm_kernel<<<dim3(H_ / NT, (B_ * S_) / MT), 256, 0, stream>>>(
        xh, wet, be, routing, out);
}
